// BernoulliRBF_86973087744579
// MI455X (gfx1250) — matrix operand order repaired and checked by reading, compiled, not run on hardware
//
#include <hip/hip_runtime.h>
#include <hip/hip_bf16.h>

// ---------------------------------------------------------------------------
// BernoulliRBF on MI455X (gfx1250, wave32).
//
// Dominant work: two [2048x256] x [256x32768] GEMMs (68.7 GFLOP) feeding a
// row-wise streaming logsumexp. Compute-bound on the matrix units (bf16 WMMA
// ~5 PFLOPS dense => ~14us/pass vs 2.8us of HBM traffic), so we use
// v_wmma_f32_16x16x32_bf16 with hi/lo split-precision compensation
// (a_hi*b_hi + a_lo*b_hi + a_hi*b_lo) for ~f32 accuracy at 3 bf16 passes.
// The 2048x32768 distance matrix is never materialized: 16x16 WMMA tiles are
// folded into per-lane online LSE state, reduced across lanes/waves/blocks.
// ---------------------------------------------------------------------------

typedef __bf16 bf16_t;
typedef __bf16 v16bf __attribute__((ext_vector_type(16)));
typedef __bf16 v8bf  __attribute__((ext_vector_type(8)));
typedef float  v8f   __attribute__((ext_vector_type(8)));

#define D_DIM   256          // feature dim (K)
#define N_X     2048         // query rows
#define M_DATA  32768        // reference rows per set
#define KT      (D_DIM / 32) // 8 k-steps of 32 per WMMA
#define GY      16           // m-chunks per set
#define NWAVES  8            // waves per block
#define NTILES  (N_X / 16)   // 128 n-tiles
#define MTILES  (M_DATA / 16)            // 2048 m-tiles per set
#define TILES_PER_BLOCK (MTILES / GY)    // 128
#define TILES_PER_WAVE  (TILES_PER_BLOCK / NWAVES) // 16

// ---------------- online / pairwise logsumexp helpers ----------------------
__device__ __forceinline__ void lse_push(float& rm, float& rs, float val) {
    float m2 = fmaxf(rm, val);
    rs = rs * __expf(rm - m2) + __expf(val - m2);
    rm = m2;
}

__device__ __forceinline__ void lse_merge(float& rm, float& rs, float m2, float s2) {
    float M = fmaxf(rm, m2);
    if (M == -__builtin_inff()) { rm = M; rs = 0.f; return; }
    rs = rs * __expf(rm - M) + s2 * __expf(m2 - M);
    rm = M;
}

// ---------------- fragment loaders (contiguous k = 16*(lane/16) + s in A and B) ----------
// A (16x32 bf16): lane<16 holds row=lane%16, K = k0..k0+15 with
// k0 = 0; lane>=16 same row, k0 = 16. Two 16B segments per lane.
__device__ __forceinline__ v16bf load_a_frag(const bf16_t* __restrict__ row_base,
                                             int lane, int kt) {
    int k0 = kt * 32 + ((lane < 16) ? 0 : 16);
    v8bf a0 = *(const v8bf*)(row_base + k0);
    v8bf a1 = *(const v8bf*)(row_base + k0 + 8);
    v16bf r;
#pragma unroll
    for (int i = 0; i < 8; ++i) { r[i] = a0[i]; r[i + 8] = a1[i]; }
    return r;
}

// B (32x16 bf16): column = lane%16 (== a data row, K contiguous in memory),
// lanes 0-15 hold K 0..15, lanes 16-31 hold K 16..31 => one 32B contiguous load.
__device__ __forceinline__ v16bf load_b_frag(const bf16_t* __restrict__ B,
                                             int mrow, int lane, int kt) {
    int k0 = kt * 32 + ((lane < 16) ? 0 : 16);
    return *(const v16bf*)(B + (size_t)mrow * D_DIM + k0);
}

// ---------------- kernel 1: scale, split f32 -> bf16 hi/lo, row norms ------
__global__ __launch_bounds__(256) void prep_rows(
    const float* __restrict__ src, const float* __restrict__ scale,
    bf16_t* __restrict__ hi, bf16_t* __restrict__ lo,
    float* __restrict__ norm) {
    __shared__ float red[256];
    int row = blockIdx.x;
    int d   = threadIdx.x;
    float v = src[(size_t)row * D_DIM + d] * scale[d];
    bf16_t h = (bf16_t)v;
    bf16_t l = (bf16_t)(v - (float)h);
    hi[(size_t)row * D_DIM + d] = h;
    lo[(size_t)row * D_DIM + d] = l;
    red[d] = v * v;
    __syncthreads();
#pragma unroll
    for (int s = 128; s > 0; s >>= 1) {
        if (d < s) red[d] += red[d + s];
        __syncthreads();
    }
    if (d == 0) norm[row] = red[0];
}

// ---------------- kernel 2: WMMA cross-term + streaming logsumexp ----------
// grid = (NTILES, GY), block = 256 (8 waves). One launch per set (pos/neg).
__global__ __launch_bounds__(256) void rbf_lse_kernel(
    const bf16_t* __restrict__ Ahi, const bf16_t* __restrict__ Alo,
    const bf16_t* __restrict__ Bhi, const bf16_t* __restrict__ Blo,
    const float* __restrict__ xnorm, const float* __restrict__ ynorm,
    const float* __restrict__ wbias,        // pre-offset to this set's bias
    float* __restrict__ pmax, float* __restrict__ psum) {
    const int ntile  = blockIdx.x;
    const int tid    = threadIdx.x;
    const int lane   = tid & 31;
    const int wave   = tid >> 5;
    const int laneN  = lane & 15;
    const int hiHalf = (lane >= 16) ? 1 : 0;

    const float bias = wbias[0];

    // A fragments for this n-tile stay resident in VGPRs (2*8*8 = 128 VGPRs).
    const bf16_t* arow_hi = Ahi + (size_t)(ntile * 16 + laneN) * D_DIM;
    const bf16_t* arow_lo = Alo + (size_t)(ntile * 16 + laneN) * D_DIM;
    v16bf a_hi[KT], a_lo[KT];
#pragma unroll
    for (int kt = 0; kt < KT; ++kt) {
        a_hi[kt] = load_a_frag(arow_hi, lane, kt);
        a_lo[kt] = load_a_frag(arow_lo, lane, kt);
    }

    // |xs|^2 for the 8 accumulator rows this lane owns (C/D layout: M=v+8*hiHalf).
    float xn[8];
#pragma unroll
    for (int v = 0; v < 8; ++v)
        xn[v] = xnorm[ntile * 16 + v + hiHalf * 8];

    float rmax[8], rsum[8];
#pragma unroll
    for (int v = 0; v < 8; ++v) { rmax[v] = -__builtin_inff(); rsum[v] = 0.f; }

    const int tile0 = blockIdx.y * TILES_PER_BLOCK + wave * TILES_PER_WAVE;

    for (int it = 0; it < TILES_PER_WAVE; it += 2) {
        const int mt0 = tile0 + it;
        const int mt1 = mt0 + 1;
        const int mrow0 = mt0 * 16 + laneN;
        const int mrow1 = mt1 * 16 + laneN;
        v8f acc0 = {}; v8f acc1 = {};
#pragma unroll
        for (int kt = 0; kt < KT; ++kt) {
            v16bf b0h = load_b_frag(Bhi, mrow0, lane, kt);
            v16bf b0l = load_b_frag(Blo, mrow0, lane, kt);
            v16bf b1h = load_b_frag(Bhi, mrow1, lane, kt);
            v16bf b1l = load_b_frag(Blo, mrow1, lane, kt);
            // split-precision dot: hi*hi + lo*hi + hi*lo (lo*lo ~ 2^-18, dropped)
            acc0 = __builtin_amdgcn_wmma_f32_16x16x32_bf16(false, a_hi[kt], false, b0h, (short)0, acc0, false, false);
            acc1 = __builtin_amdgcn_wmma_f32_16x16x32_bf16(false, a_hi[kt], false, b1h, (short)0, acc1, false, false);
            acc0 = __builtin_amdgcn_wmma_f32_16x16x32_bf16(false, a_lo[kt], false, b0h, (short)0, acc0, false, false);
            acc1 = __builtin_amdgcn_wmma_f32_16x16x32_bf16(false, a_lo[kt], false, b1h, (short)0, acc1, false, false);
            acc0 = __builtin_amdgcn_wmma_f32_16x16x32_bf16(false, a_hi[kt], false, b0l, (short)0, acc0, false, false);
            acc1 = __builtin_amdgcn_wmma_f32_16x16x32_bf16(false, a_hi[kt], false, b1l, (short)0, acc1, false, false);
        }
        const float yn0 = ynorm[mt0 * 16 + laneN];
        const float yn1 = ynorm[mt1 * 16 + laneN];
#pragma unroll
        for (int v = 0; v < 8; ++v) {
            // val = -(|xs|^2 + |ys|^2 - 2 xs.ys + bias)
            lse_push(rmax[v], rsum[v], 2.f * acc0[v] - xn[v] - yn0 - bias);
            lse_push(rmax[v], rsum[v], 2.f * acc1[v] - xn[v] - yn1 - bias);
        }
    }

    // Butterfly reduce across the 16 lanes of each half (columns of the tile).
#pragma unroll
    for (int v = 0; v < 8; ++v) {
#pragma unroll
        for (int off = 1; off < 16; off <<= 1) {
            float m2 = __shfl_xor(rmax[v], off, 32);
            float s2 = __shfl_xor(rsum[v], off, 32);
            lse_merge(rmax[v], rsum[v], m2, s2);
        }
    }

    __shared__ float smax[NWAVES][16];
    __shared__ float ssum[NWAVES][16];
    if (laneN == 0) {  // lanes 0 and 16 carry rows v and v+8 respectively
#pragma unroll
        for (int v = 0; v < 8; ++v) {
            int nl = v + hiHalf * 8;
            smax[wave][nl] = rmax[v];
            ssum[wave][nl] = rsum[v];
        }
    }
    __syncthreads();
    if (tid < 16) {
        float M = smax[0][tid], S = ssum[0][tid];
#pragma unroll
        for (int w = 1; w < NWAVES; ++w) lse_merge(M, S, smax[w][tid], ssum[w][tid]);
        int idx = (blockIdx.y * gridDim.x + blockIdx.x) * 16 + tid;
        pmax[idx] = M;
        psum[idx] = S;
    }
}

// ---------------- kernel 3: merge chunk partials, final logaddexp ----------
__global__ __launch_bounds__(256) void finalize_kernel(
    const float* __restrict__ pmaxP, const float* __restrict__ psumP,
    const float* __restrict__ pmaxN, const float* __restrict__ psumN,
    float* __restrict__ out) {
    int n = blockIdx.x * blockDim.x + threadIdx.x;
    if (n >= N_X) return;
    int ntile = n >> 4, nl = n & 15;
    float Mp = -__builtin_inff(), Sp = 0.f;
    float Mn = -__builtin_inff(), Sn = 0.f;
#pragma unroll
    for (int gy = 0; gy < GY; ++gy) {
        int idx = (gy * NTILES + ntile) * 16 + nl;
        lse_merge(Mp, Sp, pmaxP[idx], psumP[idx]);
        lse_merge(Mn, Sn, pmaxN[idx], psumN[idx]);
    }
    float lp = __logf(Sp) + Mp;                       // log_pos_count
    float ln = __logf(Sn) + Mn;                       // log_neg_count
    float mx = fmaxf(lp, ln);
    float lw = mx + __logf(__expf(lp - mx) + __expf(ln - mx)); // logaddexp
    out[n]        = lp - lw;                          // log_p_x
    out[N_X + n]  = lw;                               // log_weight_count
}

// ---------------------------------------------------------------------------
extern "C" void kernel_launch(void* const* d_in, const int* in_sizes, int n_in,
                              void* d_out, int out_size, void* d_ws, size_t ws_size,
                              hipStream_t stream) {
    const float* x          = (const float*)d_in[0];
    const float* data_pos   = (const float*)d_in[1];
    const float* data_neg   = (const float*)d_in[2];
    const float* scales_pos = (const float*)d_in[3];
    const float* scales_neg = (const float*)d_in[4];
    const float* wbias      = (const float*)d_in[5];
    float* out = (float*)d_out;

    // ---- carve workspace (~69 MB) ----
    char* ws = (char*)d_ws;
    size_t off = 0;
    auto carve = [&](size_t bytes) -> char* {
        char* p = ws + off;
        off += (bytes + 255) & ~(size_t)255;
        return p;
    };
    const size_t NA = (size_t)N_X * D_DIM;    // 524288
    const size_t NB = (size_t)M_DATA * D_DIM; // 8388608
    bf16_t* AhiP = (bf16_t*)carve(NA * 2);
    bf16_t* AloP = (bf16_t*)carve(NA * 2);
    bf16_t* AhiN = (bf16_t*)carve(NA * 2);
    bf16_t* AloN = (bf16_t*)carve(NA * 2);
    bf16_t* BhiP = (bf16_t*)carve(NB * 2);
    bf16_t* BloP = (bf16_t*)carve(NB * 2);
    bf16_t* BhiN = (bf16_t*)carve(NB * 2);
    bf16_t* BloN = (bf16_t*)carve(NB * 2);
    float* xnP = (float*)carve(N_X * 4);
    float* xnN = (float*)carve(N_X * 4);
    float* ynP = (float*)carve(M_DATA * 4);
    float* ynN = (float*)carve(M_DATA * 4);
    const size_t NPART = (size_t)GY * NTILES * 16;  // 32768 per array
    float* pmaxP = (float*)carve(NPART * 4);
    float* psumP = (float*)carve(NPART * 4);
    float* pmaxN = (float*)carve(NPART * 4);
    float* psumN = (float*)carve(NPART * 4);

    // ---- 1) preprocess: scale, bf16 hi/lo split, exact f32 row norms ----
    prep_rows<<<N_X,    256, 0, stream>>>(x,        scales_pos, AhiP, AloP, xnP);
    prep_rows<<<N_X,    256, 0, stream>>>(x,        scales_neg, AhiN, AloN, xnN);
    prep_rows<<<M_DATA, 256, 0, stream>>>(data_pos, scales_pos, BhiP, BloP, ynP);
    prep_rows<<<M_DATA, 256, 0, stream>>>(data_neg, scales_neg, BhiN, BloN, ynN);

    // ---- 2) WMMA cross terms + streaming LSE (one launch per set) ----
    dim3 grid(NTILES, GY);
    rbf_lse_kernel<<<grid, 256, 0, stream>>>(AhiP, AloP, BhiP, BloP,
                                             xnP, ynP, wbias + 0, pmaxP, psumP);
    rbf_lse_kernel<<<grid, 256, 0, stream>>>(AhiN, AloN, BhiN, BloN,
                                             xnN, ynN, wbias + 1, pmaxN, psumN);

    // ---- 3) finalize ----
    finalize_kernel<<<(N_X + 255) / 256, 256, 0, stream>>>(pmaxP, psumP,
                                                           pmaxN, psumN, out);
}
